// SimpleAttention2_60962765799921
// MI455X (gfx1250) — compile-verified, matrix operand order checked by reading, not run on hardware
//
#include <hip/hip_runtime.h>

typedef __attribute__((ext_vector_type(2)))  float  v2f;
typedef __attribute__((ext_vector_type(8)))  float  v8f;
typedef __attribute__((ext_vector_type(16))) __bf16 v16bf;

#define B_ 8
#define T_ 2048
#define S_ 2048
#define H_ 128
#define INV_TAU 1.25f   // 1/0.8
#define NTILE (S_ / 32)

// LDS carve-up (dynamic shared; > 64KB static limit, WGP has 320KB).
// All region offsets are multiples of 128 so alignment propagates.
#define QS_STRIDE   132                         // f32, conflict-free 8B loads
#define QS_FLOATS   (64 * QS_STRIDE)            // 33792 B
#define VF_STRIDE   132
#define VF_TILE     (32 * VF_STRIDE)            // floats per buffer (16896 B)
#define VT_STRIDE   36                          // bf16, conflict-free loads, 72B rows (8B aligned)
#define VT_TILE     (128 * VT_STRIDE)           // shorts per buffer (9216 B)
#define PB_STRIDE   36
#define SMEM_BYTES  (QS_FLOATS * 4 + 2 * VF_TILE * 4 + 2 * VT_TILE * 2 + 4 * 16 * PB_STRIDE * 2) // 90624

__device__ __forceinline__ unsigned short f2bf(float f) {
  unsigned u = __float_as_uint(f);
  u += 0x7FFFu + ((u >> 16) & 1u);   // round-to-nearest-even to bf16
  return (unsigned short)(u >> 16);
}

union BF16x16 { v16bf v; unsigned u[8]; };

// DPP16 lane permute (VALU, no LDS)
#define DPP_F(x, ctrl) \
  __int_as_float(__builtin_amdgcn_update_dpp(0, __float_as_int(x), (ctrl), 0xf, 0xf, true))

// Butterfly reduction within each 16-lane half of the wave:
// quad_perm(1,0,3,2)=0xB1 (xor1), quad_perm(2,3,0,1)=0x4E (xor2),
// row_mirror=0x140 (quads 0<->3,1<->2), row_half_mirror=0x141 (0<->1,2<->3).
__device__ __forceinline__ float hred_max(float x) {
  x = fmaxf(x, DPP_F(x, 0xB1));
  x = fmaxf(x, DPP_F(x, 0x4E));
  x = fmaxf(x, DPP_F(x, 0x140));
  x = fmaxf(x, DPP_F(x, 0x141));
  return x;
}
__device__ __forceinline__ float hred_sum(float x) {
  x += DPP_F(x, 0xB1);
  x += DPP_F(x, 0x4E);
  x += DPP_F(x, 0x140);
  x += DPP_F(x, 0x141);
  return x;
}

// Register-free async copy of one 32x128 f32 V tile into LDS (ASYNCcnt-tracked).
// Low 32 bits of a flat LDS pointer are the LDS byte offset (ISA 10.2 aperture map).
__device__ __forceinline__ void async_load_vtile(const float* vsrc, float* vfdst, int tid) {
  #pragma unroll
  for (int kc = 0; kc < 8; ++kc) {
    const int idx = tid + kc * 128;
    const int sr  = idx >> 5;
    const int c4  = (idx & 31) << 2;
    unsigned           lds_a = (unsigned)(uintptr_t)(vfdst + sr * VF_STRIDE + c4);
    unsigned long long ga    = (unsigned long long)(uintptr_t)(vsrc + sr * H_ + c4);
    asm volatile("global_load_async_to_lds_b128 %0, %1, off"
                 :: "v"(lds_a), "v"(ga) : "memory");
  }
}

// Fused gumbel-softmax attention, flash style, double-buffered async V pipeline:
//   per 32-col S tile: logits = Q@V^T (f32 WMMA 16x16x4), stream logits out (NT),
//   online softmax of (logits+g)/tau, P@V accumulate (bf16 WMMA 16x16x32).
// Block = 128 threads (4 wave32); each wave owns 16 T rows; block owns 64 rows.
__global__ __launch_bounds__(128)
void gumbel_attn_kernel(const float* __restrict__ q,
                        const float* __restrict__ v,
                        const float* __restrict__ g,
                        float* __restrict__ attn,
                        float* __restrict__ logits)
{
  extern __shared__ __align__(128) char smem_raw[];
  char* smem = (char*)__builtin_assume_aligned(smem_raw, 128);
  float*          Qs  = (float*)smem;                                          // [64][132]
  float*          Vfb = (float*)(smem + QS_FLOATS * 4);                        // [2][32][132]
  unsigned short* Vtb = (unsigned short*)(smem + QS_FLOATS * 4 + 2 * VF_TILE * 4); // [2][128][36]
  unsigned short* Pb  = (unsigned short*)(smem + QS_FLOATS * 4 + 2 * VF_TILE * 4 + 2 * VT_TILE * 2);

  const int tid  = threadIdx.x;
  const int wid  = tid >> 5;
  const int lane = tid & 31;
  const int lh   = lane & 15;   // column / M-low index inside fragments
  const int hs   = lane >> 4;   // half-wave select

  const int    b  = blockIdx.y;
  const int    t0 = blockIdx.x * 64 + wid * 16;
  const size_t bt = (size_t)b * T_;

  // ---- prologue: async-load V tile 0; stage Q tile (64x128) into LDS ----
  async_load_vtile(v + (size_t)b * S_ * H_, Vfb, tid);
  {
    const float* qptr = q + (bt + (size_t)blockIdx.x * 64) * H_;
    #pragma unroll
    for (int kc = 0; kc < 16; ++kc) {
      const int idx = tid + kc * 128;
      const int r   = idx >> 5;
      const int c4  = (idx & 31) << 2;
      const float4 val = *(const float4*)(qptr + r * H_ + c4);
      float* dst = Qs + r * QS_STRIDE + c4;
      dst[0] = val.x; dst[1] = val.y; dst[2] = val.z; dst[3] = val.w;
    }
  }
  asm volatile("s_wait_asynccnt 0" ::: "memory");
  __syncthreads();

  // ---- accumulators: out tile 16x128 = 8 C-frags; per-row online-softmax stats ----
  v8f co[8];
  float m_i[8], l_i[8];
  #pragma unroll
  for (int j = 0; j < 8; ++j) {
    #pragma unroll
    for (int f = 0; f < 8; ++f) co[f][j] = 0.0f;
    m_i[j] = -3.0e38f;
    l_i[j] = 0.0f;
  }

  for (int st = 0; st < NTILE; ++st) {
    const int s0  = st * 32;
    const int cur = st & 1;
    float*          Vf = Vfb + cur * VF_TILE;
    unsigned short* Vt = Vtb + cur * VT_TILE;

    // issue async copy of tile st+1 into the other buffer (consumers of it
    // finished before the barrier that ended iteration st-1)
    if (st + 1 < NTILE)
      async_load_vtile(v + ((size_t)b * S_ + s0 + 32) * H_, Vfb + (cur ^ 1) * VF_TILE, tid);

    // build bf16 transposed copy Vt[cur] from Vf[cur] (LDS->LDS via registers)
    #pragma unroll
    for (int kc = 0; kc < 8; ++kc) {
      const int idx = tid + kc * 128;
      const int sr  = idx >> 5;
      const int c4  = (idx & 31) << 2;
      const float4 val = *(const float4*)(Vf + sr * VF_STRIDE + c4);
      Vt[(c4 + 0) * VT_STRIDE + sr] = f2bf(val.x);
      Vt[(c4 + 1) * VT_STRIDE + sr] = f2bf(val.y);
      Vt[(c4 + 2) * VT_STRIDE + sr] = f2bf(val.z);
      Vt[(c4 + 3) * VT_STRIDE + sr] = f2bf(val.w);
    }

    // ---- QK^T: 16 rows x 32 cols, K=H=128 via 64x V_WMMA_F32_16X16X4_F32 ----
    v8f z0, z1;
    #pragma unroll
    for (int j = 0; j < 8; ++j) { z0[j] = 0.0f; z1[j] = 0.0f; }
    {
      const float* qrow = Qs + (wid * 16 + lh) * QS_STRIDE;
      #pragma unroll
      for (int kk = 0; kk < 32; ++kk) {
        const int k0 = kk * 4 + hs * 2;          // lanes 0-15: K=0,1 ; 16-31: K=2,3
        const v2f a  = *(const v2f*)(qrow + k0);
        const v2f b0 = *(const v2f*)(Vf + lh * VF_STRIDE + k0);        // B[k][n]=V[s0+n][k]
        const v2f b1 = *(const v2f*)(Vf + (16 + lh) * VF_STRIDE + k0);
        z0 = __builtin_amdgcn_wmma_f32_16x16x4_f32(false, a, false, b0, (short)0, z0, false, false);
        z1 = __builtin_amdgcn_wmma_f32_16x16x4_f32(false, a, false, b1, (short)0, z1, false, false);
      }
    }

    // ---- stream logits out (pre-noise, NT), fetch gumbel (NT), scale by 1/tau ----
    float z0s[8], z1s[8];
    #pragma unroll
    for (int j = 0; j < 8; ++j) {
      const int    t    = t0 + j + 8 * hs;       // C-frag row for VGPR j
      const size_t base = (bt + t) * (size_t)S_ + s0;
      __builtin_nontemporal_store(z0[j], &logits[base + lh]);
      __builtin_nontemporal_store(z1[j], &logits[base + 16 + lh]);
      const float g0 = __builtin_nontemporal_load(&g[base + lh]);
      const float g1 = __builtin_nontemporal_load(&g[base + 16 + lh]);
      z0s[j] = (z0[j] + g0) * INV_TAU;
      z1s[j] = (z1[j] + g1) * INV_TAU;
    }
    if (st + 1 < NTILE) {   // L2 prefetch of next tile's gumbel (global_prefetch_b8)
      __builtin_prefetch(&g[(bt + t0 + lh) * (size_t)S_ + s0 + 32 + hs * 16], 0, 1);
    }

    // ---- online softmax update (DPP16 reductions); P -> bf16 into LDS A-frag staging ----
    #pragma unroll
    for (int j = 0; j < 8; ++j) {
      const float tm   = hred_max(fmaxf(z0s[j], z1s[j]));   // row max across 16 lanes
      const float mn   = fmaxf(m_i[j], tm);
      const float corr = __expf(m_i[j] - mn);
      m_i[j] = mn;
      const float p0 = __expf(z0s[j] - mn);
      const float p1 = __expf(z1s[j] - mn);
      l_i[j] = l_i[j] * corr + hred_sum(p0 + p1);
      #pragma unroll
      for (int f = 0; f < 8; ++f) co[f][j] *= corr;  // rescale matches frag row layout
      Pb[(wid * 16 + j + 8 * hs) * PB_STRIDE + lh]      = f2bf(p0);
      Pb[(wid * 16 + j + 8 * hs) * PB_STRIDE + 16 + lh] = f2bf(p1);
    }

    __syncthreads();                               // B2: Vt[cur] published by all waves
    asm volatile("s_wait_dscnt 0" ::: "memory");   // own P staging stores landed

    // ---- P@V: 8x V_WMMA_F32_16X16X32_BF16 into out accumulators ----
    BF16x16 pa;
    {
      const unsigned short* prow = Pb + (wid * 16 + lh) * PB_STRIDE;
      #pragma unroll
      for (int r = 0; r < 8; ++r) {
        const int k = (r & 3) * 2 + hs * 8 + (r >> 2) * 16;   // bf16 A-frag K pairs
        pa.u[r] = *(const unsigned*)(prow + k);
      }
    }
    #pragma unroll
    for (int f = 0; f < 8; ++f) {
      const unsigned short* vrow = Vt + (f * 16 + lh) * VT_STRIDE;
      BF16x16 vbf;
      #pragma unroll
      for (int r = 0; r < 8; ++r) {
        const int k = (r & 3) * 2 + hs * 8 + (r >> 2) * 16;   // bf16 B-frag K pairs
        vbf.u[r] = *(const unsigned*)(vrow + k);              // B[k][n]=V[s0+k][16f+n]
      }
      co[f] = __builtin_amdgcn_wmma_f32_16x16x32_bf16(false, pa.v, false, vbf.v,
                                                      (short)0, co[f], false, false);
    }

    // B1: my async stores for tile st+1 landed; everyone done with tile st
    asm volatile("s_wait_asynccnt 0" ::: "memory");
    __syncthreads();
  }

  // ---- finalize: divide by row sums, write attn ----
  #pragma unroll
  for (int j = 0; j < 8; ++j) {
    const int   t   = t0 + j + 8 * hs;
    const float inv = 1.0f / l_i[j];
    float* orow = attn + (bt + t) * (size_t)H_;
    #pragma unroll
    for (int f = 0; f < 8; ++f)
      orow[f * 16 + lh] = co[f][j] * inv;
  }
}

extern "C" void kernel_launch(void* const* d_in, const int* in_sizes, int n_in,
                              void* d_out, int out_size, void* d_ws, size_t ws_size,
                              hipStream_t stream) {
  const float* q = (const float*)d_in[0];
  const float* v = (const float*)d_in[1];
  const float* g = (const float*)d_in[2];
  float* attn   = (float*)d_out;                              // [B,T,H]
  float* logits = (float*)d_out + (size_t)B_ * T_ * H_;       // [B,T,S]
  dim3 grid(T_ / 64, B_);
  gumbel_attn_kernel<<<grid, 128, SMEM_BYTES, stream>>>(q, v, g, attn, logits);
}
